// MultiHeadAttention_19980187861467
// MI455X (gfx1250) — matrix operand order repaired and checked by reading, not compiled, not run on hardware
//
#include <hip/hip_runtime.h>
#include <hip/hip_bf16.h>
#include <stdint.h>

#define SEQ   2048
#define EMB   1024
#define NH    16
#define HD    64
#define BATCH 2
#define MROWS (BATCH * SEQ) // 4096

typedef __attribute__((ext_vector_type(16))) __bf16 v16bf;
typedef __attribute__((ext_vector_type(8)))  float  v8f;

union FragBF { unsigned int u[8]; v16bf v; };

__device__ __forceinline__ unsigned short f2bf(float x) {
    unsigned int u = __float_as_uint(x);
    unsigned int r = u + 0x7fffu + ((u >> 16) & 1u); // round-to-nearest-even
    return (unsigned short)(r >> 16);
}

// A-fragment (16x32 bf16) K offset for packed-pair j (0..7), half hi (0/1)
__device__ __forceinline__ int koffA(int j, int hi) {
    return (j < 4 ? 2 * j : 2 * j + 8) + 8 * hi;
}
// B-fragment (32x16 bf16) K offset
__device__ __forceinline__ int koffB(int j, int hi) {
    return (j < 4 ? 2 * j : 2 * j + 8) + 8 * hi;
}

// Async copy 16 B global -> LDS (per lane), tracked by ASYNCcnt.
__device__ __forceinline__ void async_g2l_b128(void* lds_ptr, const void* gptr) {
    unsigned lds_off = (unsigned)(size_t)lds_ptr;     // low 32 bits = LDS offset
    unsigned long long ga = (unsigned long long)(size_t)gptr;
    asm volatile("global_load_async_to_lds_b128 %0, %1, off"
                 :: "v"(lds_off), "v"(ga) : "memory");
}
__device__ __forceinline__ void wait_async0() {
    asm volatile("s_wait_asynccnt 0x0" ::: "memory");
}

// ---------------------------------------------------------------- convert
__global__ void cvt_f32_bf16(const float* __restrict__ src,
                             unsigned short* __restrict__ dst, int n) {
    int i = (blockIdx.x * blockDim.x + threadIdx.x) * 4;
    if (i + 3 < n) {
        float4 v = *(const float4*)(src + i);
        dst[i + 0] = f2bf(v.x);
        dst[i + 1] = f2bf(v.y);
        dst[i + 2] = f2bf(v.z);
        dst[i + 3] = f2bf(v.w);
    }
}

// ---------------------------------------------------------------- QKV GEMM
// C[4096,1024] = X[4096,1024] @ W[1024,1024] + b ; scatter to [B,H,S,hd] bf16
__global__ __launch_bounds__(256) void qkv_gemm(
    const unsigned short* __restrict__ Xb,
    const unsigned short* __restrict__ Wqb, const unsigned short* __restrict__ Wkb,
    const unsigned short* __restrict__ Wvb,
    const float* __restrict__ bq, const float* __restrict__ bk, const float* __restrict__ bv,
    unsigned short* __restrict__ Qb, unsigned short* __restrict__ Kb,
    unsigned short* __restrict__ Vb) {
    __shared__ __align__(16) unsigned short As[64 * 40];  // 80 B rows (16B aligned)
    __shared__ __align__(16) unsigned short Bs[128 * 34]; // [n][k] transposed

    const int tid = threadIdx.x;
    const int lane = tid & 31, w = tid >> 5;
    const int lr = lane & 15, hi = lane >> 4;
    const int m0 = blockIdx.y * 64;
    const int n0 = blockIdx.x * 128;
    const int waveM = (w & 1) * 32;
    const int waveN = (w >> 1) * 32;

    const unsigned short* W;
    const float* bias;
    unsigned short* out;
    if (blockIdx.z == 0)      { W = Wqb; bias = bq; out = Qb; }
    else if (blockIdx.z == 1) { W = Wkb; bias = bk; out = Kb; }
    else                      { W = Wvb; bias = bv; out = Vb; }

    const v8f zf = {0.f, 0.f, 0.f, 0.f, 0.f, 0.f, 0.f, 0.f};
    v8f acc[2][2];
    acc[0][0] = zf; acc[0][1] = zf; acc[1][0] = zf; acc[1][1] = zf;

    const int rA = tid >> 2, cA = (tid & 3) * 8;     // A staging: 8 bf16/thread
    const int kkB = tid >> 3, nbB = (tid & 7) * 16;  // B staging: 16 bf16/thread

    for (int k0 = 0; k0 < EMB; k0 += 32) {
        // A tile 64x32 row-major: async global->LDS, no VGPR round trip
        async_g2l_b128(&As[rA * 40 + cA],
                       Xb + (size_t)(m0 + rA) * EMB + k0 + cA);
        { // B tile 32x128 -> transposed [n][k] in LDS
            const unsigned short* gp = W + (size_t)(k0 + kkB) * EMB + n0 + nbB;
            unsigned short tmp[16];
            *(uint4*)tmp       = *(const uint4*)gp;
            *(uint4*)(tmp + 8) = *(const uint4*)(gp + 8);
#pragma unroll
            for (int i = 0; i < 16; ++i) Bs[(nbB + i) * 34 + kkB] = tmp[i];
        }
        wait_async0();
        __syncthreads();

        FragBF a[2], b[2];
#pragma unroll
        for (int mi = 0; mi < 2; ++mi) {
            int row = waveM + mi * 16 + lr;
#pragma unroll
            for (int j = 0; j < 8; ++j)
                a[mi].u[j] = *(const unsigned int*)&As[row * 40 + koffA(j, hi)];
        }
#pragma unroll
        for (int ni = 0; ni < 2; ++ni) {
            int col = waveN + ni * 16 + lr;
#pragma unroll
            for (int j = 0; j < 8; ++j)
                b[ni].u[j] = *(const unsigned int*)&Bs[col * 34 + koffB(j, hi)];
        }
#pragma unroll
        for (int mi = 0; mi < 2; ++mi)
#pragma unroll
            for (int ni = 0; ni < 2; ++ni)
                acc[mi][ni] = __builtin_amdgcn_wmma_f32_16x16x32_bf16(
                    false, a[mi].v, false, b[ni].v, (short)0, acc[mi][ni], false, false);
        __syncthreads();
    }

#pragma unroll
    for (int mi = 0; mi < 2; ++mi)
#pragma unroll
        for (int ni = 0; ni < 2; ++ni) {
            int n = n0 + waveN + ni * 16 + lr;
            float bv_ = bias[n];
            int h = n >> 6, d = n & 63;
#pragma unroll
            for (int v = 0; v < 8; ++v) {
                int gm = m0 + waveM + mi * 16 + v + 8 * hi;
                int bb = gm >> 11, s = gm & 2047;
                out[((size_t)(bb * NH + h) * SEQ + s) * HD + d] =
                    f2bf(acc[mi][ni][v] + bv_);
            }
        }
}

// ---------------------------------------------------------------- attention
__global__ __launch_bounds__(256) void attn_kernel(
    const unsigned short* __restrict__ Qb, const unsigned short* __restrict__ Kb,
    const unsigned short* __restrict__ Vb, unsigned short* __restrict__ Ab) {
    __shared__ __align__(16) unsigned short Kt[32 * 72]; // [key][hd], 144 B rows
    __shared__ __align__(16) unsigned short Vt[64 * 34]; // [hd][key] transposed
    __shared__ __align__(16) float          Sld[8][16 * 32];
    __shared__ __align__(16) unsigned short Pld[8][16 * 32];
    __shared__ float Mld[8][16], Lld[8][16], Ald[8][16];

    const int tid = threadIdx.x;
    const int lane = tid & 31, w = tid >> 5;
    const int lr = lane & 15, hi = lane >> 4;
    const int bh = blockIdx.x;            // b*NH + h
    const int Q0 = blockIdx.y * 128;
    const int q0w = Q0 + w * 16;          // this wave's first query row
    const size_t baseBH = (size_t)bh * SEQ * HD;

    const v8f zf = {0.f, 0.f, 0.f, 0.f, 0.f, 0.f, 0.f, 0.f};

    // Q fragments for this wave's 16 rows (hd chunks 0-31, 32-63)
    FragBF qf[2];
#pragma unroll
    for (int kc = 0; kc < 2; ++kc) {
        const unsigned short* qp = Qb + baseBH + (size_t)(q0w + lr) * HD + kc * 32;
#pragma unroll
        for (int j = 0; j < 8; ++j)
            qf[kc].u[j] = *(const unsigned int*)(qp + koffA(j, hi));
    }

    v8f o[4];
    o[0] = zf; o[1] = zf; o[2] = zf; o[3] = zf;
    if (hi == 0) { Mld[w][lr] = -1.0e30f; Lld[w][lr] = 0.0f; }

    const int keyS = tid >> 3, cS = (tid & 7) * 8;  // staging: 8 bf16/thread

    const int ntiles = Q0 / 32 + 4; // covers causal extent of this q block
    for (int t = 0; t < ntiles; ++t) {
        const int kv0 = t * 32;
        // K tile: contiguous -> async global->LDS
        async_g2l_b128(&Kt[keyS * 72 + cS],
                       Kb + baseBH + (size_t)(kv0 + keyS) * HD + cS);
        { // V tile: transpose to [hd][key]
            uint4 dv = *(const uint4*)(Vb + baseBH + (size_t)(kv0 + keyS) * HD + cS);
            unsigned short tmp[8];
            *(uint4*)tmp = dv;
#pragma unroll
            for (int i = 0; i < 8; ++i) Vt[(cS + i) * 34 + keyS] = tmp[i];
        }
        wait_async0();
        __syncthreads();

        if (kv0 <= q0w + 15) { // wave-uniform causal skip (EXEC stays full)
            // S = Q @ K^T  (16 x 32)
            v8f sacc[2];
            sacc[0] = zf; sacc[1] = zf;
#pragma unroll
            for (int ni = 0; ni < 2; ++ni) {
                int key = ni * 16 + lr;
#pragma unroll
                for (int kc = 0; kc < 2; ++kc) {
                    FragBF bfr;
#pragma unroll
                    for (int j = 0; j < 8; ++j)
                        bfr.u[j] = *(const unsigned int*)&Kt[key * 72 + kc * 32 + koffB(j, hi)];
                    sacc[ni] = __builtin_amdgcn_wmma_f32_16x16x32_bf16(
                        false, qf[kc].v, false, bfr.v, (short)0, sacc[ni], false, false);
                }
            }
            // scale + causal mask, spill to LDS
#pragma unroll
            for (int ni = 0; ni < 2; ++ni) {
                int col = ni * 16 + lr;
                int gk = kv0 + col;
#pragma unroll
                for (int v = 0; v < 8; ++v) {
                    int row = v + 8 * hi;
                    float val = sacc[ni][v] * 0.125f; // 1/sqrt(64)
                    if (q0w + row < gk) val = -10000.0f;
                    Sld[w][row * 32 + col] = val;
                }
            }
            // online softmax: lanes 0..15 each own one row
            if (hi == 0) {
                int r = lr;
                float mo = Mld[w][r], lo = Lld[w][r];
                float mx = mo;
#pragma unroll
                for (int j = 0; j < 32; ++j) mx = fmaxf(mx, Sld[w][r * 32 + j]);
                float alpha = __expf(mo - mx);
                float sum = 0.f;
#pragma unroll
                for (int j = 0; j < 32; ++j) {
                    float p = __expf(Sld[w][r * 32 + j] - mx);
                    sum += p;
                    Pld[w][r * 32 + j] = f2bf(p);
                }
                Mld[w][r] = mx;
                Lld[w][r] = lo * alpha + sum;
                Ald[w][r] = alpha;
            }
            // rescale running O
#pragma unroll
            for (int i = 0; i < 4; ++i)
#pragma unroll
                for (int v = 0; v < 8; ++v) o[i][v] *= Ald[w][v + 8 * hi];
            // P fragment (16x32 bf16)
            FragBF pf;
#pragma unroll
            for (int j = 0; j < 8; ++j)
                pf.u[j] = *(const unsigned int*)&Pld[w][lr * 32 + koffA(j, hi)];
            // O += P @ V   (16 x 64, contracted over 32 keys)
#pragma unroll
            for (int ni = 0; ni < 4; ++ni) {
                int n = ni * 16 + lr;
                FragBF vfr;
#pragma unroll
                for (int j = 0; j < 8; ++j)
                    vfr.u[j] = *(const unsigned int*)&Vt[n * 34 + koffB(j, hi)];
                o[ni] = __builtin_amdgcn_wmma_f32_16x16x32_bf16(
                    false, pf.v, false, vfr.v, (short)0, o[ni], false, false);
            }
        }
        __syncthreads();
    }

    // normalize and write [B,S,D] bf16
    const int b = bh >> 4, h = bh & 15;
#pragma unroll
    for (int ni = 0; ni < 4; ++ni) {
        int hd = ni * 16 + lr;
#pragma unroll
        for (int v = 0; v < 8; ++v) {
            int row = v + 8 * hi;
            float l = Lld[w][row];
            float val = o[ni][v] / l;
            Ab[((size_t)b * SEQ + (q0w + row)) * EMB + h * HD + hd] = f2bf(val);
        }
    }
}

// ---------------------------------------------------------------- out GEMM
__global__ __launch_bounds__(256) void out_gemm(
    const unsigned short* __restrict__ Abuf, const unsigned short* __restrict__ Wob,
    const float* __restrict__ bo, float* __restrict__ out) {
    __shared__ __align__(16) unsigned short As[64 * 40];
    __shared__ __align__(16) unsigned short Bs[128 * 34];

    const int tid = threadIdx.x;
    const int lane = tid & 31, w = tid >> 5;
    const int lr = lane & 15, hi = lane >> 4;
    const int m0 = blockIdx.y * 64;
    const int n0 = blockIdx.x * 128;
    const int waveM = (w & 1) * 32;
    const int waveN = (w >> 1) * 32;

    const v8f zf = {0.f, 0.f, 0.f, 0.f, 0.f, 0.f, 0.f, 0.f};
    v8f acc[2][2];
    acc[0][0] = zf; acc[0][1] = zf; acc[1][0] = zf; acc[1][1] = zf;

    const int rA = tid >> 2, cA = (tid & 3) * 8;
    const int kkB = tid >> 3, nbB = (tid & 7) * 16;

    for (int k0 = 0; k0 < EMB; k0 += 32) {
        async_g2l_b128(&As[rA * 40 + cA],
                       Abuf + (size_t)(m0 + rA) * EMB + k0 + cA);
        {
            const unsigned short* gp = Wob + (size_t)(k0 + kkB) * EMB + n0 + nbB;
            unsigned short tmp[16];
            *(uint4*)tmp       = *(const uint4*)gp;
            *(uint4*)(tmp + 8) = *(const uint4*)(gp + 8);
#pragma unroll
            for (int i = 0; i < 16; ++i) Bs[(nbB + i) * 34 + kkB] = tmp[i];
        }
        wait_async0();
        __syncthreads();

        FragBF a[2], b[2];
#pragma unroll
        for (int mi = 0; mi < 2; ++mi) {
            int row = waveM + mi * 16 + lr;
#pragma unroll
            for (int j = 0; j < 8; ++j)
                a[mi].u[j] = *(const unsigned int*)&As[row * 40 + koffA(j, hi)];
        }
#pragma unroll
        for (int ni = 0; ni < 2; ++ni) {
            int col = waveN + ni * 16 + lr;
#pragma unroll
            for (int j = 0; j < 8; ++j)
                b[ni].u[j] = *(const unsigned int*)&Bs[col * 34 + koffB(j, hi)];
        }
#pragma unroll
        for (int mi = 0; mi < 2; ++mi)
#pragma unroll
            for (int ni = 0; ni < 2; ++ni)
                acc[mi][ni] = __builtin_amdgcn_wmma_f32_16x16x32_bf16(
                    false, a[mi].v, false, b[ni].v, (short)0, acc[mi][ni], false, false);
        __syncthreads();
    }

#pragma unroll
    for (int mi = 0; mi < 2; ++mi)
#pragma unroll
        for (int ni = 0; ni < 2; ++ni) {
            int n = n0 + waveN + ni * 16 + lr;
            float bv_ = bo[n];
#pragma unroll
            for (int v = 0; v < 8; ++v) {
                int gm = m0 + waveM + mi * 16 + v + 8 * hi;
                out[(size_t)gm * EMB + n] = acc[mi][ni][v] + bv_;
            }
        }
}

// ---------------------------------------------------------------- launch
extern "C" void kernel_launch(void* const* d_in, const int* in_sizes, int n_in,
                              void* d_out, int out_size, void* d_ws, size_t ws_size,
                              hipStream_t stream) {
    (void)in_sizes; (void)n_in; (void)out_size; (void)ws_size;
    const float* X  = (const float*)d_in[0];
    const float* Wq = (const float*)d_in[1];
    const float* bq = (const float*)d_in[2];
    const float* Wk = (const float*)d_in[3];
    const float* bk = (const float*)d_in[4];
    const float* Wv = (const float*)d_in[5];
    const float* bv = (const float*)d_in[6];
    const float* Wo = (const float*)d_in[7];
    const float* bo = (const float*)d_in[8];
    float* out = (float*)d_out;

    char* ws = (char*)d_ws;
    const size_t SZ_X = (size_t)MROWS * EMB * 2; // 8 MB
    const size_t SZ_W = (size_t)EMB * EMB * 2;   // 2 MB
    unsigned short* Xb   = (unsigned short*)(ws);
    unsigned short* Wqb  = (unsigned short*)(ws + SZ_X);
    unsigned short* Wkb  = (unsigned short*)(ws + SZ_X + SZ_W);
    unsigned short* Wvb  = (unsigned short*)(ws + SZ_X + 2 * SZ_W);
    unsigned short* Wob  = (unsigned short*)(ws + SZ_X + 3 * SZ_W);
    unsigned short* Qb   = (unsigned short*)(ws + SZ_X + 4 * SZ_W);
    unsigned short* Kbuf = (unsigned short*)(ws + 2 * SZ_X + 4 * SZ_W);
    unsigned short* Vbuf = (unsigned short*)(ws + 3 * SZ_X + 4 * SZ_W);
    unsigned short* Abuf = (unsigned short*)(ws + 4 * SZ_X + 4 * SZ_W);

    int nX = MROWS * EMB, nW = EMB * EMB;
    cvt_f32_bf16<<<(nX / 4 + 255) / 256, 256, 0, stream>>>(X, Xb, nX);
    cvt_f32_bf16<<<(nW / 4 + 255) / 256, 256, 0, stream>>>(Wq, Wqb, nW);
    cvt_f32_bf16<<<(nW / 4 + 255) / 256, 256, 0, stream>>>(Wk, Wkb, nW);
    cvt_f32_bf16<<<(nW / 4 + 255) / 256, 256, 0, stream>>>(Wv, Wvb, nW);
    cvt_f32_bf16<<<(nW / 4 + 255) / 256, 256, 0, stream>>>(Wo, Wob, nW);

    dim3 g1(EMB / 128, MROWS / 64, 3);
    qkv_gemm<<<g1, 256, 0, stream>>>(Xb, Wqb, Wkb, Wvb, bq, bk, bv, Qb, Kbuf, Vbuf);

    dim3 g2(BATCH * NH, SEQ / 128);
    attn_kernel<<<g2, 256, 0, stream>>>(Qb, Kbuf, Vbuf, Abuf);

    dim3 g3(EMB / 128, MROWS / 64);
    out_gemm<<<g3, 256, 0, stream>>>(Abuf, Wob, bo, out);
}
